// BayesLSTM_52166672777898
// MI455X (gfx1250) — matrix operand order repaired and checked by reading, not compiled, not run on hardware
//
#include <hip/hip_runtime.h>
#include <hip/hip_bf16.h>
#include <stdint.h>

#define B_  64
#define T_  512
#define D_  512
#define H_  1024
#define G4H (4 * H_)

typedef __bf16 bf16_t;
typedef __attribute__((ext_vector_type(8)))  bf16_t bf16x8;
typedef __attribute__((ext_vector_type(16))) bf16_t bf16x16;
typedef __attribute__((ext_vector_type(8)))  float  f32x8;

union AF { bf16x16 v; bf16x8 h[2]; };

__device__ __forceinline__ unsigned short f2bf(float f) {
    unsigned u = __float_as_uint(f);
    u += 0x7FFFu + ((u >> 16) & 1u);           // round-to-nearest-even
    return (unsigned short)(u >> 16);
}
__device__ __forceinline__ float sigf(float x)     { return 1.0f / (1.0f + __expf(-x)); }
__device__ __forceinline__ float tanhfast(float x) { return 2.0f / (1.0f + __expf(-2.0f * x)) - 1.0f; }

// ---- grid-wide barrier (persistent kernel; 64 co-resident workgroups) ----
__device__ __forceinline__ void gsync(unsigned* cnt, unsigned* gen, unsigned nblk) {
    __threadfence();
    __syncthreads();
    if (threadIdx.x == 0) {
        unsigned g = __hip_atomic_load(gen, __ATOMIC_ACQUIRE, __HIP_MEMORY_SCOPE_AGENT);
        unsigned a = __hip_atomic_fetch_add(cnt, 1u, __ATOMIC_ACQ_REL, __HIP_MEMORY_SCOPE_AGENT) + 1u;
        if (a == nblk) {
            __hip_atomic_store(cnt, 0u, __ATOMIC_RELAXED, __HIP_MEMORY_SCOPE_AGENT);
            __hip_atomic_fetch_add(gen, 1u, __ATOMIC_RELEASE, __HIP_MEMORY_SCOPE_AGENT);
        } else {
            while (__hip_atomic_load(gen, __ATOMIC_ACQUIRE, __HIP_MEMORY_SCOPE_AGENT) == g) {
                __builtin_amdgcn_s_sleep(1);
            }
        }
    }
    __syncthreads();
}

// ---- setup: x -> time-major bf16, in_mask applied at t==0 ----
__global__ void k_prep_x(const float* __restrict__ x, const float* __restrict__ in_mask,
                         unsigned short* __restrict__ xbf) {
    const size_t N = (size_t)T_ * B_ * D_;
    const size_t stride = (size_t)gridDim.x * blockDim.x;
    for (size_t i = (size_t)blockIdx.x * blockDim.x + threadIdx.x; i < N; i += stride) {
        int t   = (int)(i >> 15);      // / (B*D)
        int rem = (int)(i & 32767);
        int b   = rem >> 9;            // / D
        int d   = rem & 511;
        float v = x[((size_t)b * T_ + t) * D_ + d];
        if (t == 0) v *= in_mask[b * D_ + d];
        xbf[i] = f2bf(v);
    }
}

// ---- setup: weights -> bf16, bias combine ----
__global__ void k_prep_w(const float* __restrict__ wih_f, const float* __restrict__ whh_f,
                         const float* __restrict__ b_ih, const float* __restrict__ b_hh,
                         unsigned short* __restrict__ wih, unsigned short* __restrict__ whh,
                         float* __restrict__ bias) {
    const size_t stride = (size_t)gridDim.x * blockDim.x;
    const size_t i0 = (size_t)blockIdx.x * blockDim.x + threadIdx.x;
    const size_t NI = (size_t)G4H * D_;
    const size_t NH = (size_t)G4H * H_;
    for (size_t k = i0; k < NI;  k += stride) wih[k] = f2bf(wih_f[k]);
    for (size_t k = i0; k < NH;  k += stride) whh[k] = f2bf(whh_f[k]);
    for (size_t k = i0; k < G4H; k += stride) bias[k] = b_ih[k] + b_hh[k];
}

// ---- setup: zero h/c state and barrier words ----
__global__ void k_init(unsigned short* __restrict__ hbf, float* __restrict__ cst,
                       unsigned* __restrict__ bar) {
    int i = blockIdx.x * blockDim.x + threadIdx.x;
    if (i < B_ * H_) { hbf[i] = 0; cst[i] = 0.0f; }
    if (i < 64) bar[i] = 0u;
}

// ---- persistent recurrent kernel: 64 blocks x 256 threads = 512 waves ----
// Each wave owns 1 M-tile x 2 adjacent N-tiles (A fragment reused twice).
__global__ __launch_bounds__(256)
void lstm_persist(const unsigned short* __restrict__ xbf,
                  const unsigned short* __restrict__ wih,
                  const unsigned short* __restrict__ whh,
                  const float* __restrict__ bias,
                  float* __restrict__ gates,
                  unsigned short* __restrict__ hbf,
                  float* __restrict__ cst,
                  const float* __restrict__ hid_mask,
                  const float* __restrict__ out_mask,
                  float* __restrict__ y, float* __restrict__ hf, float* __restrict__ cf,
                  unsigned* bar_cnt, unsigned* bar_gen) {
    const int lane = threadIdx.x & 31;
    const int wave = (blockIdx.x << 3) + (threadIdx.x >> 5);   // 0..511
    const int m0 = (wave & 3) << 4;                            // 0,16,32,48
    const int n0 = (wave >> 2) << 5;                           // 0..4064 step 32
    const int lr = lane & 15;
    const int hi = lane >> 4;
    const float bv0 = bias[n0 + lr];
    const float bv1 = bias[n0 + 16 + lr];
    const int col0  = n0 + lr;
    const int rbase = m0 + (hi << 3);

    // B fragments: lane holds column (n0+lr); 32 contiguous bf16 of W row, split by hi.
    const bf16_t* wB0  = (const bf16_t*)wih + (size_t)(n0 + lr)      * D_ + (hi << 4);
    const bf16_t* wB0b = (const bf16_t*)wih + (size_t)(n0 + 16 + lr) * D_ + (hi << 4);
    const bf16_t* wB1  = (const bf16_t*)whh + (size_t)(n0 + lr)      * H_ + (hi << 4);
    const bf16_t* wB1b = (const bf16_t*)whh + (size_t)(n0 + 16 + lr) * H_ + (hi << 4);
    // A fragment base for h (row = batch m0+lr, 16 contiguous elems by hi).
    const bf16_t* hA = (const bf16_t*)hbf + (size_t)(m0 + lr) * H_ + (hi << 4);
    const int tidg = blockIdx.x * 256 + threadIdx.x;           // 16384 threads

    for (int t = 0; t < T_; ++t) {
        f32x8 acc0, acc1;
        #pragma unroll
        for (int r = 0; r < 8; ++r) { acc0[r] = bv0; acc1[r] = bv1; }

        const bf16_t* xA = (const bf16_t*)xbf + (size_t)t * (B_ * D_)
                         + (size_t)(m0 + lr) * D_ + (hi << 4);
        #pragma unroll 4
        for (int k = 0; k < D_; k += 32) {                     // x @ W_ih^T
            AF a; a.h[0] = *(const bf16x8*)(xA + k); a.h[1] = *(const bf16x8*)(xA + k + 8);
            bf16x16 b0 = *(const bf16x16*)(wB0  + k);
            bf16x16 b1 = *(const bf16x16*)(wB0b + k);
            acc0 = __builtin_amdgcn_wmma_f32_16x16x32_bf16(false, a.v, false, b0, (short)0, acc0, false, false);
            acc1 = __builtin_amdgcn_wmma_f32_16x16x32_bf16(false, a.v, false, b1, (short)0, acc1, false, false);
        }
        #pragma unroll 4
        for (int k = 0; k < H_; k += 32) {                     // h @ W_hh^T
            AF a; a.h[0] = *(const bf16x8*)(hA + k); a.h[1] = *(const bf16x8*)(hA + k + 8);
            bf16x16 b0 = *(const bf16x16*)(wB1  + k);
            bf16x16 b1 = *(const bf16x16*)(wB1b + k);
            acc0 = __builtin_amdgcn_wmma_f32_16x16x32_bf16(false, a.v, false, b0, (short)0, acc0, false, false);
            acc1 = __builtin_amdgcn_wmma_f32_16x16x32_bf16(false, a.v, false, b1, (short)0, acc1, false, false);
        }
        #pragma unroll
        for (int r = 0; r < 8; ++r) {                          // C/D layout: M=r+8*hi, N=lane%16
            gates[(size_t)(rbase + r) * G4H + col0]      = acc0[r];
            gates[(size_t)(rbase + r) * G4H + col0 + 16] = acc1[r];
        }

        gsync(bar_cnt, bar_gen, gridDim.x);

        // activations: 65536 (b,j) elems over 16384 threads
        #pragma unroll
        for (int it = 0; it < 4; ++it) {
            int e = tidg + it * 16384;
            int b = e >> 10, j = e & (H_ - 1);
            const float* gr = gates + (size_t)b * G4H;
            float ig = sigf(gr[j]);
            float fg = sigf(gr[H_ + j]);
            float gg = tanhfast(gr[2 * H_ + j]);
            float og = sigf(gr[3 * H_ + j]);
            float cn = fg * cst[e] + ig * gg;
            cst[e] = cn;
            float hn = og * tanhfast(cn);
            y[((size_t)b * T_ + t) * H_ + j] = hn * out_mask[e];
            float hm = hn * hid_mask[e];
            hbf[e] = f2bf(hm);
            if (t == T_ - 1) { hf[e] = hm; cf[e] = cn; }
        }

        gsync(bar_cnt, bar_gen, gridDim.x);
    }
}

// ---- workspace layout (bytes) ----
#define WS_XBF   ((size_t)0)                       // T*B*D*2      = 33,554,432
#define WS_WIH   ((size_t)33554432)                // 4H*D*2       =  4,194,304
#define WS_WHH   ((size_t)37748736)                // 4H*H*2       =  8,388,608
#define WS_BIAS  ((size_t)46137344)                // 4H*4         =     16,384
#define WS_GATES ((size_t)46153728)                // B*4H*4       =  1,048,576
#define WS_HBF   ((size_t)47202304)                // B*H*2        =    131,072
#define WS_CST   ((size_t)47333376)                // B*H*4        =    262,144
#define WS_BAR   ((size_t)47595520)                // 256

extern "C" void kernel_launch(void* const* d_in, const int* in_sizes, int n_in,
                              void* d_out, int out_size, void* d_ws, size_t ws_size,
                              hipStream_t stream) {
    const float* x        = (const float*)d_in[0];
    const float* W_ih     = (const float*)d_in[1];
    const float* W_hh     = (const float*)d_in[2];
    const float* b_ih     = (const float*)d_in[3];
    const float* b_hh     = (const float*)d_in[4];
    const float* in_mask  = (const float*)d_in[5];
    const float* hid_mask = (const float*)d_in[6];
    const float* out_mask = (const float*)d_in[7];

    char* ws = (char*)d_ws;
    unsigned short* xbf  = (unsigned short*)(ws + WS_XBF);
    unsigned short* wih  = (unsigned short*)(ws + WS_WIH);
    unsigned short* whh  = (unsigned short*)(ws + WS_WHH);
    float*          bias = (float*)(ws + WS_BIAS);
    float*          gts  = (float*)(ws + WS_GATES);
    unsigned short* hbf  = (unsigned short*)(ws + WS_HBF);
    float*          cst  = (float*)(ws + WS_CST);
    unsigned*       bar  = (unsigned*)(ws + WS_BAR);

    float* y  = (float*)d_out;                     // [B,T,H]
    float* hf = y + (size_t)B_ * T_ * H_;          // [B,H]
    float* cf = hf + (size_t)B_ * H_;              // [B,H]

    k_prep_x<<<8192, 256, 0, stream>>>(x, in_mask, xbf);
    k_prep_w<<<2048, 256, 0, stream>>>(W_ih, W_hh, b_ih, b_hh, wih, whh, bias);
    k_init<<<256, 256, 0, stream>>>(hbf, cst, bar);
    lstm_persist<<<64, 256, 0, stream>>>(xbf, wih, whh, bias, gts, hbf, cst,
                                         hid_mask, out_mask, y, hf, cf,
                                         bar, bar + 1);
}
